// OFAAttention_55327768707691
// MI455X (gfx1250) — matrix operand order repaired and checked by reading, not compiled, not run on hardware
//
#include <hip/hip_runtime.h>
#include <hip/hip_bf16.h>

// ---------------------------------------------------------------------------
// OFA attention for MI455X (gfx1250, wave32, WMMA bf16 + async LDS copies)
//   k1: gemm_xwT<1> (X fp32 * W^T fp32 + b) -> bf16 (b,h,t,d)  [Q,K,V]
//   k2: flash_attn fused QK^T + bias + mask + online softmax + PV
//   k3: gemm_xwT<0> final projection -> d_out fp32
// ---------------------------------------------------------------------------

typedef __attribute__((ext_vector_type(16))) __bf16 v16bf;
typedef __attribute__((ext_vector_type(8)))  __bf16 v8bf;
typedef __attribute__((ext_vector_type(8)))  float  v8f;
typedef __attribute__((ext_vector_type(4)))  int    v4i;

#define B_SZ 2
#define N_HEADS 16
#define HEAD_D 64
#define EMB 1024
#define TGT 2048
#define M_ROWS (B_SZ * TGT)              // 4096
#define SCALING_F 0.08838834764831845f   // (64*2)^-0.5

// ---- async memory->LDS copy (ASYNCcnt path), with safe fallback ----------
#if defined(__HIP_DEVICE_COMPILE__) && __has_builtin(__builtin_amdgcn_global_load_async_to_lds_b128)
#define ASYNC_COPY 1
#endif

__device__ __forceinline__ void copy16_to_lds(void* lds_dst, const void* gsrc) {
#ifdef ASYNC_COPY
  typedef __attribute__((address_space(1))) v4i* gptr_t;   // global int4*
  typedef __attribute__((address_space(3))) v4i* lptr_t;   // LDS int4*
  __builtin_amdgcn_global_load_async_to_lds_b128(
      (gptr_t)gsrc, (lptr_t)lds_dst, /*imm offset=*/0, /*cpol=*/0);
#else
  *(v8bf*)lds_dst = *(const v8bf*)gsrc;
#endif
}

__device__ __forceinline__ void async_copy_wait() {
#ifdef ASYNC_COPY
#if __has_builtin(__builtin_amdgcn_s_wait_asynccnt)
  __builtin_amdgcn_s_wait_asynccnt(0);
#else
  asm volatile("s_wait_asynccnt 0x0" ::: "memory");
#endif
#endif
}

// ---- WMMA helpers ---------------------------------------------------------
__device__ __forceinline__ v8f wmma_bf16(v16bf a, v16bf b, v8f c) {
  return __builtin_amdgcn_wmma_f32_16x16x32_bf16(
      /*neg_a=*/false, a, /*neg_b=*/false, b,
      /*c_mod=*/(short)0, c, /*reuse_a=*/false, /*reuse_b=*/false);
}

// A fragment (16xK32, bf16): lanes 0-15 row=lane, K = {kb..kb+7, kb+16..kb+23}
//                            lanes16-31 row=lane-16, K = {kb+8..15, kb+24..31}
__device__ __forceinline__ v16bf load_frag_a(const __bf16* p0, int ld, int lane) {
  const __bf16* p = p0 + (lane & 15) * ld + ((lane >> 4) << 3);
  v8bf lo = *(const v8bf*)p;
  v8bf hi = *(const v8bf*)(p + 16);
  v16bf r;
#pragma unroll
  for (int i = 0; i < 8; ++i) { r[i] = lo[i]; r[i + 8] = hi[i]; }
  return r;
}

// B fragment (K32x16, bf16): lanes 0-15 col=lane, K = {kb..kb+7, kb+16..kb+23}
//                            lanes16-31 col=lane-16, K = {kb+8..15, kb+24..31}
__device__ __forceinline__ v16bf load_frag_b(const __bf16* p0, int ld, int lane) {
  const __bf16* p = p0 + (lane & 15) * ld + ((lane >> 4) << 3);
  v8bf lo = *(const v8bf*)p;
  v8bf hi = *(const v8bf*)(p + 16);
  v16bf r;
#pragma unroll
  for (int i = 0; i < 8; ++i) { r[i] = lo[i]; r[i + 8] = hi[i]; }
  return r;
}

// ---------------------------------------------------------------------------
// C = (X[M,K] * W[N,K]^T + bias[N]) * alpha
// MODE 0: outF[M,N] fp32 row-major
// MODE 1: outB bf16 at (b, h, t, d) with row=b*TGT+t, col=h*64+d
// Tile: 64(M) x 128(N), K-step 32; 256 threads = 8 waves (2x4), 32x32 each.
// ---------------------------------------------------------------------------
template <int MODE>
__global__ __launch_bounds__(256)
void gemm_xwT(const float* __restrict__ X, const float* __restrict__ W,
              const float* __restrict__ bias, int M, int N, int K,
              float alpha,
              float* __restrict__ outF, __bf16* __restrict__ outB) {
  __shared__ __attribute__((aligned(16))) __bf16 ldsA[64 * 32];
  __shared__ __attribute__((aligned(16))) __bf16 ldsB[128 * 32];

  const int tid  = threadIdx.x;
  const int lane = tid & 31;
  const int wave = tid >> 5;
  const int wm = wave & 1;        // 0..1 -> M offset wm*32
  const int wn = wave >> 1;       // 0..3 -> N offset wn*32
  const int m0 = blockIdx.y * 64;
  const int n0 = blockIdx.x * 128;

  v8f acc[2][2] = {};

  for (int k0 = 0; k0 < K; k0 += 32) {
    __syncthreads();
    // A tile 64x32 (fp32 -> bf16)
    for (int i = tid; i < 64 * 8; i += 256) {
      const int r = i >> 3, c = (i & 7) << 2;
      const float4 x = *(const float4*)(X + (size_t)(m0 + r) * K + k0 + c);
      ldsA[r * 32 + c + 0] = (__bf16)x.x;
      ldsA[r * 32 + c + 1] = (__bf16)x.y;
      ldsA[r * 32 + c + 2] = (__bf16)x.z;
      ldsA[r * 32 + c + 3] = (__bf16)x.w;
    }
    // B tile 128x32 (rows of W = columns of W^T)
    for (int i = tid; i < 128 * 8; i += 256) {
      const int r = i >> 3, c = (i & 7) << 2;
      const float4 x = *(const float4*)(W + (size_t)(n0 + r) * K + k0 + c);
      ldsB[r * 32 + c + 0] = (__bf16)x.x;
      ldsB[r * 32 + c + 1] = (__bf16)x.y;
      ldsB[r * 32 + c + 2] = (__bf16)x.z;
      ldsB[r * 32 + c + 3] = (__bf16)x.w;
    }
    __syncthreads();

    v16bf af[2], bf2[2];
#pragma unroll
    for (int mt = 0; mt < 2; ++mt)
      af[mt] = load_frag_a(ldsA + (wm * 32 + mt * 16) * 32, 32, lane);
#pragma unroll
    for (int nt = 0; nt < 2; ++nt)
      bf2[nt] = load_frag_b(ldsB + (wn * 32 + nt * 16) * 32, 32, lane);
#pragma unroll
    for (int mt = 0; mt < 2; ++mt)
#pragma unroll
      for (int nt = 0; nt < 2; ++nt)
        acc[mt][nt] = wmma_bf16(af[mt], bf2[nt], acc[mt][nt]);
  }

  // Epilogue. C layout: VGPR j -> row (j or 8+j), col = lane&15 per half.
  const int rhalf = (lane >> 4) << 3;
#pragma unroll
  for (int mt = 0; mt < 2; ++mt) {
#pragma unroll
    for (int nt = 0; nt < 2; ++nt) {
      const int rowb = m0 + wm * 32 + mt * 16 + rhalf;
      const int col  = n0 + wn * 32 + nt * 16 + (lane & 15);
      const float bc = bias[col];
      const v8f c = acc[mt][nt];
#pragma unroll
      for (int j = 0; j < 8; ++j) {
        const float v = (c[j] + bc) * alpha;
        const int r = rowb + j;
        if (MODE == 0) {
          outF[(size_t)r * N + col] = v;
        } else {
          const int bb = r >> 11, t = r & (TGT - 1);
          const int hh = col >> 6, dd = col & 63;
          outB[((((size_t)bb * N_HEADS + hh) * TGT + t) << 6) + dd] = (__bf16)v;
        }
      }
    }
  }
}

// ---------------------------------------------------------------------------
// Fused flash attention. Grid: (TGT/64, B*H), block 128 (4 waves).
// Wave w owns q rows [q0+16w, q0+16w+16). Streams 64-key tiles.
// Q/K tiles are contiguous 8KB blocks -> async global->LDS copies.
// ---------------------------------------------------------------------------
__global__ __launch_bounds__(128)
void flash_attn(const __bf16* __restrict__ Q, const __bf16* __restrict__ Km,
                const __bf16* __restrict__ V, const float* __restrict__ bias,
                const float* __restrict__ mask, const float* __restrict__ c_attn,
                float* __restrict__ out) {
  __shared__ __attribute__((aligned(16))) __bf16 ldsQ[64 * 64];
  __shared__ __attribute__((aligned(16))) __bf16 ldsK[64 * 64];
  __shared__ __attribute__((aligned(16))) __bf16 ldsVT[64 * 64];
  __shared__ __attribute__((aligned(16))) __bf16 ldsP[4 * 16 * 64];

  const int tid = threadIdx.x, lane = tid & 31, wave = tid >> 5;
  const int bh = blockIdx.y, b = bh >> 4, h = bh & 15;
  const int q0 = blockIdx.x * 64;
  const int rhalf = (lane >> 4) << 3;
  const int qrow0 = q0 + wave * 16 + rhalf;

  // Q tile: one contiguous 8KB block
  const __bf16* Qb = Q + ((size_t)bh * TGT + q0) * HEAD_D;
  for (int i = tid; i < 512; i += 128)
    copy16_to_lds(ldsQ + i * 8, Qb + (size_t)i * 8);

  v8f Oacc[4] = {};
  float mrow[8], lrow[8];
#pragma unroll
  for (int j = 0; j < 8; ++j) { mrow[j] = -1e30f; lrow[j] = 0.f; }
  __bf16* ldsPw = ldsP + wave * 16 * 64;

  async_copy_wait();
  __syncthreads();

  for (int s0 = 0; s0 < TGT; s0 += 64) {
    const __bf16* Kb = Km + ((size_t)bh * TGT + s0) * HEAD_D;
    const __bf16* Vb = V + ((size_t)bh * TGT + s0) * HEAD_D;
    // K tile: contiguous 8KB -> async copy
    for (int i = tid; i < 512; i += 128)
      copy16_to_lds(ldsK + i * 8, Kb + (size_t)i * 8);
    // V tile transposed into VT[d][s]
    for (int i = tid; i < 64 * 64; i += 128) {
      const int r = i >> 6, c = i & 63;
      ldsVT[c * 64 + r] = Vb[(size_t)r * HEAD_D + c];
    }
    async_copy_wait();
    __syncthreads();

    // Prefetch next s-tile's bias rows (dominant HBM stream: 537 MB)
    if (s0 + 64 < TGT) {
      const float* bpn = bias + ((size_t)bh * TGT + qrow0) * TGT + (s0 + 64) + (lane & 15);
#pragma unroll
      for (int j = 0; j < 8; ++j) {
        __builtin_prefetch(bpn + (size_t)j * TGT, 0, 1);
        __builtin_prefetch(bpn + (size_t)j * TGT + 32, 0, 1);
      }
    }

    // S = Q * K^T  (16 q-rows x 64 keys per wave)
    v8f S[4] = {};
    const v16bf qa0 = load_frag_a(ldsQ + wave * 16 * 64 + 0, 64, lane);
    const v16bf qa1 = load_frag_a(ldsQ + wave * 16 * 64 + 32, 64, lane);
#pragma unroll
    for (int nt = 0; nt < 4; ++nt) {
      const v16bf kb0 = load_frag_b(ldsK + nt * 16 * 64 + 0, 64, lane);
      const v16bf kb1 = load_frag_b(ldsK + nt * 16 * 64 + 32, 64, lane);
      S[nt] = wmma_bf16(qa0, kb0, S[nt]);
      S[nt] = wmma_bf16(qa1, kb1, S[nt]);
    }

    // + attn_bias + attention_mask (loaded straight into the C layout)
#pragma unroll
    for (int nt = 0; nt < 4; ++nt) {
      const int col = s0 + nt * 16 + (lane & 15);
      const float* bp = bias + ((size_t)bh * TGT + qrow0) * TGT + col;
      const float* mp = mask + ((size_t)b * TGT + qrow0) * TGT + col;
#pragma unroll
      for (int j = 0; j < 8; ++j)
        S[nt][j] += bp[(size_t)j * TGT] + mp[(size_t)j * TGT];
    }

    // online softmax (row spread over 16 lanes of a half; xor-reduce 1,2,4,8)
#pragma unroll
    for (int j = 0; j < 8; ++j) {
      float mx = fmaxf(fmaxf(S[0][j], S[1][j]), fmaxf(S[2][j], S[3][j]));
      mx = fmaxf(mx, __shfl_xor(mx, 1, 32));
      mx = fmaxf(mx, __shfl_xor(mx, 2, 32));
      mx = fmaxf(mx, __shfl_xor(mx, 4, 32));
      mx = fmaxf(mx, __shfl_xor(mx, 8, 32));
      const float mnew = fmaxf(mrow[j], mx);
      const float corr = __expf(mrow[j] - mnew);
      float s = 0.f;
#pragma unroll
      for (int nt = 0; nt < 4; ++nt) {
        const float p = __expf(S[nt][j] - mnew);
        S[nt][j] = p;
        s += p;
      }
      s += __shfl_xor(s, 1, 32);
      s += __shfl_xor(s, 2, 32);
      s += __shfl_xor(s, 4, 32);
      s += __shfl_xor(s, 8, 32);
      lrow[j] = lrow[j] * corr + s;
      mrow[j] = mnew;
#pragma unroll
      for (int dt = 0; dt < 4; ++dt) Oacc[dt][j] *= corr;
    }

    // P -> bf16 A fragments via per-wave LDS (no cross-wave barrier needed)
#pragma unroll
    for (int nt = 0; nt < 4; ++nt)
#pragma unroll
      for (int j = 0; j < 8; ++j)
        ldsPw[(rhalf + j) * 64 + nt * 16 + (lane & 15)] = (__bf16)S[nt][j];

    const v16bf pa0 = load_frag_a(ldsPw + 0, 64, lane);
    const v16bf pa1 = load_frag_a(ldsPw + 32, 64, lane);
#pragma unroll
    for (int dt = 0; dt < 4; ++dt) {
      const v16bf vb0 = load_frag_b(ldsVT + dt * 16 * 64 + 0, 64, lane);
      const v16bf vb1 = load_frag_b(ldsVT + dt * 16 * 64 + 32, 64, lane);
      Oacc[dt] = wmma_bf16(pa0, vb0, Oacc[dt]);
      Oacc[dt] = wmma_bf16(pa1, vb1, Oacc[dt]);
    }
    __syncthreads();
  }

  // epilogue: * c_attn[h] / l, write (b, t, h*64+d) fp32
  const float ca = c_attn[h];
#pragma unroll
  for (int j = 0; j < 8; ++j) {
    const float inv = ca / lrow[j];
    const int t = q0 + wave * 16 + rhalf + j;
#pragma unroll
    for (int dt = 0; dt < 4; ++dt) {
      const int d = dt * 16 + (lane & 15);
      out[((size_t)b * TGT + t) * EMB + h * HEAD_D + d] = Oacc[dt][j] * inv;
    }
  }
}

// ---------------------------------------------------------------------------
extern "C" void kernel_launch(void* const* d_in, const int* in_sizes, int n_in,
                              void* d_out, int out_size, void* d_ws, size_t ws_size,
                              hipStream_t stream) {
  const float* hidden = (const float*)d_in[0];
  const float* attn_bias = (const float*)d_in[1];
  const float* attn_mask = (const float*)d_in[2];
  const float* Wq = (const float*)d_in[3];
  const float* bq = (const float*)d_in[4];
  const float* Wk = (const float*)d_in[5];
  const float* bk = (const float*)d_in[6];
  const float* Wv = (const float*)d_in[7];
  const float* bv = (const float*)d_in[8];
  const float* Wo = (const float*)d_in[9];
  const float* bo = (const float*)d_in[10];
  const float* c_attn = (const float*)d_in[11];

  // workspace layout
  __bf16* qb = (__bf16*)d_ws;                         // 4096*1024 bf16
  __bf16* kb = qb + (size_t)M_ROWS * EMB;
  __bf16* vb = kb + (size_t)M_ROWS * EMB;
  float*  attn = (float*)(vb + (size_t)M_ROWS * EMB); // 4096*1024 fp32

  dim3 gg(EMB / 128, M_ROWS / 64);   // (8, 64)
  gemm_xwT<1><<<gg, 256, 0, stream>>>(hidden, Wq, bq, M_ROWS, EMB, EMB,
                                      SCALING_F, nullptr, qb);
  gemm_xwT<1><<<gg, 256, 0, stream>>>(hidden, Wk, bk, M_ROWS, EMB, EMB,
                                      1.0f, nullptr, kb);
  gemm_xwT<1><<<gg, 256, 0, stream>>>(hidden, Wv, bv, M_ROWS, EMB, EMB,
                                      1.0f, nullptr, vb);

  flash_attn<<<dim3(TGT / 64, B_SZ * N_HEADS), 128, 0, stream>>>(
      qb, kb, vb, attn_bias, attn_mask, c_attn, attn);

  gemm_xwT<0><<<gg, 256, 0, stream>>>(attn, Wo, bo, M_ROWS, EMB, EMB,
                                      1.0f, (float*)d_out, nullptr);
}
